// GemNetT_P3M_57904749085214
// MI455X (gfx1250) — matrix operand order repaired and checked by reading, not compiled, not run on hardware
//
#include <hip/hip_runtime.h>
#include <cstdint>
#include <cstddef>

// ============================================================================
// GemNet-T + P3M hybrid message passing for MI455X (gfx1250, wave32, WMMA).
//
// All dense layers run through one WMMA GEMM kernel:
//   - f32 inputs converted to f16 while staging into LDS, f32 accumulation
//     via v_wmma_f32_16x16x32_f16 (16x16x32), 8 waves/block, 128x64 tile.
//   - Staging is vectorized: float4 global reads, packed ds_store_b128 writes;
//     guards and gather indices hoisted out of the K loop.
//   - A-source modes: direct, 2-way row gather (W_comb), concat3 (W_cat).
//   - fused epilogues: +bias, SiLU, elementwise-mul (rbf modulation), +res.
// Scatter stages (triplet aggregation, rbf_h-gated edge->atom, CFConv) use
// global f32 atomics; their working sets (<52MB) are L2-resident (192MB L2).
// LayerNorm uses one wave32 per row with __shfl_xor reductions.
// MHA uses K/V tiles in LDS (64KB) with online softmax (dh=16, trivial FLOPs).
//
// Workspace requirement: 130,560,000 floats = 522,240,000 bytes.
// Params dict assumed flattened in JAX sorted-key pytree order.
// ============================================================================

typedef __attribute__((ext_vector_type(16))) _Float16 v16h;
typedef __attribute__((ext_vector_type(8)))  float    v8f;

union HFrag { uint4 u[2]; v16h h; };
union PK16  { _Float16 h[16]; uint4 u[2]; };
union PK8   { _Float16 h[8];  uint4 u; };

static constexpr int NATOMS = 10000;
static constexpr int NMESH  = 32768;
static constexpr int NE     = 200000;
static constexpr int NT     = 1500000;
static constexpr int EA2M   = 80000;

__device__ __forceinline__ float dsilu(float x) {
    return x * (1.0f / (1.0f + __expf(-x)));
}

// ---------------------------------------------------------------------------
// Generic WMMA GEMM: out[M,N] = epi(A[M,K] @ B[K,N])
// amode: 0 = A direct (ld=K)
//        1 = gather2: k<128 -> A[gis[r]*128+k], else A[git[r]*128+k-128] (K=256)
//        2 = cat3:    k<128 -> A[gis[r]], k<256 -> A[git[r]], else A2[r*256+..]
//            (segment boundaries are multiples of 16, so a thread's 16-wide
//             k-segment never straddles them)
// flags: 1=+bias[col], 2=SiLU, 4=*mul[r,c], 8=+res[r,c]
// ---------------------------------------------------------------------------
__global__ __launch_bounds__(256) void k_gemm(
    const float* __restrict__ A, const float* __restrict__ A2,
    const int* __restrict__ gis, const int* __restrict__ git,
    const float* __restrict__ B, const float* __restrict__ bias,
    const float* __restrict__ mul, const float* __restrict__ res,
    float* __restrict__ out, int M, int N, int K, int amode, int flags)
{
    __shared__ _Float16 lA[128 * 32];   // [row][k]
    __shared__ _Float16 lB[64 * 32];    // [n][k]  (k-contiguous per column)
    const int tid  = threadIdx.x;
    const int bm   = blockIdx.x * 128;
    const int bn   = blockIdx.y * 64;
    const int w    = tid >> 5;
    const int lane = tid & 31;

    // --- per-thread staging ownership (hoisted out of the K loop) ----------
    // A: thread owns row (tid>>1), k-segment of 16 at (tid&1)*16
    const int arow = tid >> 1;
    const int akk0 = (tid & 1) * 16;
    const int grow = bm + arow;
    const bool rowOK = (grow < M);
    const float* pS = nullptr;   // k in [0,128) (or full row for amode 0)
    const float* pT = nullptr;   // k in [128,256)
    const float* pU = nullptr;   // k in [256,640) (amode 2)
    if (rowOK) {
        if (amode == 0) {
            pS = A + (size_t)grow * K;
        } else {
            pS = A + (size_t)gis[grow] * 128;
            pT = A + (size_t)git[grow] * 128;
            if (amode == 2) pU = A2 + (size_t)grow * 256;
        }
    }
    // B: thread owns column (tid&63), k-segment of 8 at (tid>>6)*8
    const int bcol = tid & 63;
    const int bk0  = (tid >> 6) * 8;

    v8f acc[4];
#pragma unroll
    for (int s = 0; s < 4; ++s)
#pragma unroll
        for (int r = 0; r < 8; ++r) acc[s][r] = 0.0f;

    const int nck = (K + 31) >> 5;
    for (int ck = 0; ck < nck; ++ck) {
        const int kc = ck << 5;
        // ---- stage A segment: 16 floats -> 16 halves -> 2x ds_store_b128
        {
            PK16 pk;
            const int kb = kc + akk0;
            if (!rowOK) {
#pragma unroll
                for (int j = 0; j < 16; ++j) pk.h[j] = (_Float16)0.0f;
            } else if (amode == 0) {
                if (kb + 16 <= K) {
                    const float4* s4 = (const float4*)(pS + kb);
#pragma unroll
                    for (int j = 0; j < 4; ++j) {
                        const float4 f = s4[j];
                        pk.h[4 * j + 0] = (_Float16)f.x;
                        pk.h[4 * j + 1] = (_Float16)f.y;
                        pk.h[4 * j + 2] = (_Float16)f.z;
                        pk.h[4 * j + 3] = (_Float16)f.w;
                    }
                } else {
#pragma unroll
                    for (int j = 0; j < 16; ++j) {
                        const int k = kb + j;
                        pk.h[j] = (_Float16)((k < K) ? pS[k] : 0.0f);
                    }
                }
            } else {
                const float* seg;
                int off;
                if (kb < 128)      { seg = pS; off = kb; }
                else if (kb < 256) { seg = pT; off = kb - 128; }
                else               { seg = pU; off = kb - 256; }
                const float4* s4 = (const float4*)(seg + off);
#pragma unroll
                for (int j = 0; j < 4; ++j) {
                    const float4 f = s4[j];
                    pk.h[4 * j + 0] = (_Float16)f.x;
                    pk.h[4 * j + 1] = (_Float16)f.y;
                    pk.h[4 * j + 2] = (_Float16)f.z;
                    pk.h[4 * j + 3] = (_Float16)f.w;
                }
            }
            uint4* dA = (uint4*)&lA[arow * 32 + akk0];
            dA[0] = pk.u[0];
            dA[1] = pk.u[1];
        }
        // ---- stage B segment: 8 coalesced dword reads -> 1x ds_store_b128
        {
            PK8 p;
            const int kb = kc + bk0;
            const float* pB = B + (size_t)kb * N + bn + bcol;
            if (kc + 32 <= K) {
#pragma unroll
                for (int j = 0; j < 8; ++j)
                    p.h[j] = (_Float16)pB[(size_t)j * N];
            } else {
#pragma unroll
                for (int j = 0; j < 8; ++j)
                    p.h[j] = (_Float16)((kb + j < K) ? pB[(size_t)j * N] : 0.0f);
            }
            *(uint4*)&lB[bcol * 32 + bk0] = p.u;
        }
        __syncthreads();

        // A fragment: lane m = lane&15, K-half kh (ISA 16-bit A 16x32 layout)
        HFrag a;
        const int mrow = lane & 15, kh = lane >> 4;
        const uint4* pa = (const uint4*)&lA[(w * 16 + mrow) * 32];
        a.u[0] = pa[kh];       // K = kh*8 .. kh*8+7
        a.u[1] = pa[2 + kh];   // K = 16+kh*8 .. 16+kh*8+7
#pragma unroll
        for (int s = 0; s < 4; ++s) {
            HFrag bf;
            const uint4* pb = (const uint4*)&lB[(s * 16 + mrow) * 32];
            bf.u[0] = pb[kh];       // K = kh*8 .. kh*8+7
            bf.u[1] = pb[2 + kh];   // K = 16+kh*8 .. 16+kh*8+7
            acc[s] = __builtin_amdgcn_wmma_f32_16x16x32_f16(
                false, a.h, false, bf.h, (short)0, acc[s], false, false);
        }
        __syncthreads();
    }

    // epilogue + store (C/D layout: VGPR r -> M = r + 8*(lane>=16), N = lane&15)
    const int rbase = bm + w * 16 + (lane >> 4) * 8;
    const int c0 = bn + (lane & 15);
#pragma unroll
    for (int s = 0; s < 4; ++s) {
        const int col = c0 + s * 16;
#pragma unroll
        for (int r = 0; r < 8; ++r) {
            const int row = rbase + r;
            if (row < M) {
                float v = acc[s][r];
                if (flags & 1) v += bias[col];
                if (flags & 2) v = dsilu(v);
                if (flags & 4) v *= mul[(size_t)row * N + col];
                if (flags & 8) v += res[(size_t)row * N + col];
                out[(size_t)row * N + col] = v;
            }
        }
    }
}

// ---------------------------------------------------------------------------
__global__ void k_zero(float* __restrict__ p, long long n) {
    long long i = (long long)blockIdx.x * blockDim.x + threadIdx.x;
    const long long st = (long long)gridDim.x * blockDim.x;
    for (; i < n; i += st) p[i] = 0.0f;
}

// LayerNorm: one wave32 per row, 8 rows per 256-thread block.
__global__ __launch_bounds__(256) void k_ln(
    const float* __restrict__ in, const float* __restrict__ g,
    const float* __restrict__ b, float* __restrict__ out, int rows, int C)
{
    const int lane = threadIdx.x & 31;
    const int row  = blockIdx.x * 8 + (threadIdx.x >> 5);
    if (row >= rows) return;
    const float* x = in + (size_t)row * C;
    float s = 0.0f, ss = 0.0f;
    for (int c = lane; c < C; c += 32) { float v = x[c]; s += v; ss += v * v; }
    for (int o = 16; o; o >>= 1) { s += __shfl_xor(s, o, 32); ss += __shfl_xor(ss, o, 32); }
    const float mu = s / C;
    const float rs = rsqrtf(ss / C - mu * mu + 1e-5f);
    float* y = out + (size_t)row * C;
    for (int c = lane; c < C; c += 32) y[c] = (x[c] - mu) * rs * g[c] + b[c];
}

// out1[row] = m_x[row] + mxo[row] + LN(msg[row])   (C = 128)
__global__ __launch_bounds__(256) void k_out2(
    const float* __restrict__ msg, const float* __restrict__ mx0,
    const float* __restrict__ mxo, const float* __restrict__ g,
    const float* __restrict__ b, float* __restrict__ out, int rows)
{
    const int lane = threadIdx.x & 31;
    const int row  = blockIdx.x * 8 + (threadIdx.x >> 5);
    if (row >= rows) return;
    const float* x = msg + (size_t)row * 128;
    float s = 0.0f, ss = 0.0f;
    for (int c = lane; c < 128; c += 32) { float v = x[c]; s += v; ss += v * v; }
    for (int o = 16; o; o >>= 1) { s += __shfl_xor(s, o, 32); ss += __shfl_xor(ss, o, 32); }
    const float mu = s * (1.0f / 128.0f);
    const float rs = rsqrtf(ss * (1.0f / 128.0f) - mu * mu + 1e-5f);
    for (int c = lane; c < 128; c += 32) {
        const size_t i = (size_t)row * 128 + c;
        out[i] = mx0[i] + mxo[i] + (x[c] - mu) * rs * g[c] + b[c];
    }
}

// out3[row] += LN(em[row])   (C = 256)
__global__ __launch_bounds__(256) void k_out3(
    const float* __restrict__ em, const float* __restrict__ g,
    const float* __restrict__ b, float* __restrict__ out, int rows)
{
    const int lane = threadIdx.x & 31;
    const int row  = blockIdx.x * 8 + (threadIdx.x >> 5);
    if (row >= rows) return;
    const float* x = em + (size_t)row * 256;
    float s = 0.0f, ss = 0.0f;
    for (int c = lane; c < 256; c += 32) { float v = x[c]; s += v; ss += v * v; }
    for (int o = 16; o; o >>= 1) { s += __shfl_xor(s, o, 32); ss += __shfl_xor(ss, o, 32); }
    const float mu = s * (1.0f / 256.0f);
    const float rs = rsqrtf(ss * (1.0f / 256.0f) - mu * mu + 1e-5f);
    for (int c = lane; c < 256; c += 32) {
        const size_t i = (size_t)row * 256 + c;
        out[i] += (x[c] - mu) * rs * g[c] + b[c];
    }
}

// Triplet stage: x3p[id3_ca[t]] += x_down[id3_ba[t]] * (cbf3[t] @ W_cbf)
// 4 triplets per 256-thread block; W_cbf (16x64) and cbf3 rows in LDS.
__global__ __launch_bounds__(256) void k_triplet(
    const float* __restrict__ cbf3, const float* __restrict__ Wc,
    const float* __restrict__ xdown, const int* __restrict__ ba,
    const int* __restrict__ ca, float* __restrict__ x3p, int T)
{
    __shared__ float lw[16 * 64];
    __shared__ float lc[4 * 16];
    const int tid = threadIdx.x;
    const int tb  = blockIdx.x * 4;
    for (int i = tid; i < 1024; i += 256) lw[i] = Wc[i];
    if (tid < 64) {
        const int tt = tb + (tid >> 4);
        const int i  = tid & 15;
        lc[tid] = (tt < T) ? cbf3[(size_t)tt * 16 + i] : 0.0f;
    }
    __syncthreads();
    const int tl = tid >> 6;   // triplet within block (0..3)
    const int c  = tid & 63;   // channel (TRIP = 64)
    const int t  = tb + tl;
    if (t >= T) return;
    float s = 0.0f;
#pragma unroll
    for (int i = 0; i < 16; ++i) s += lc[tl * 16 + i] * lw[i * 64 + c];
    const float val = xdown[(size_t)ba[t] * 64 + c] * s;
    atomicAdd(&x3p[(size_t)ca[t] * 64 + c], val);
}

// m_new[e] = m_ln[e] + x3[e] + x3[id_swap[e]]   (C = 256)
__global__ void k_mnew(const float* __restrict__ mln, const float* __restrict__ x3,
                       const int* __restrict__ sw, float* __restrict__ mnew)
{
    const int e = blockIdx.x, c = threadIdx.x;
    const size_t i = (size_t)e * 256 + c;
    mnew[i] = mln[i] + x3[i] + x3[(size_t)sw[e] * 256 + c];
}

// h[idx_t[e]] += m_new[e] * gate[e]   (C = 256)
__global__ void k_hagg(const float* __restrict__ mnew, const float* __restrict__ gate,
                       const int* __restrict__ dst, float* __restrict__ h)
{
    const int e = blockIdx.x, c = threadIdx.x;
    const size_t i = (size_t)e * 256 + c;
    atomicAdd(&h[(size_t)dst[e] * 256 + c], mnew[i] * gate[i]);
}

// CFConv edge scatter: agg[dst[e]] += xin[src[e]] * filt[e] * cutoff(w[e])  (C=128)
__global__ void k_cfconv(const float* __restrict__ xin, const float* __restrict__ filt,
                         const float* __restrict__ w, const int* __restrict__ src,
                         const int* __restrict__ dst, float* __restrict__ agg, int E)
{
    const int gid = blockIdx.x * blockDim.x + threadIdx.x;
    const int e = gid >> 7, c = gid & 127;
    if (e >= E) return;
    const float cw = 0.5f * (__cosf(3.14159265358979f * w[e] * (1.0f / 3.0f)) + 1.0f);
    const float v  = xin[(size_t)src[e] * 128 + c] * filt[(size_t)e * 128 + c] * cw;
    atomicAdd(&agg[(size_t)dst[e] * 128 + c], v);
}

// MHA: block per (graph, head); K/V head tiles (512x16 each) in LDS;
// online softmax, 2 query rows per thread.
__global__ __launch_bounds__(256) void k_attn(
    const float* __restrict__ Q, const float* __restrict__ K,
    const float* __restrict__ V, float* __restrict__ O)
{
    __shared__ float sK[512 * 16];
    __shared__ float sV[512 * 16];
    const int g = blockIdx.x >> 3, h = blockIdx.x & 7;
    const int tid = threadIdx.x;
    const size_t base = ((size_t)g * 512) * 128 + (size_t)h * 16;
    for (int i = tid; i < 512 * 16; i += 256) {
        const int tok = i >> 4, d = i & 15;
        sK[i] = K[base + (size_t)tok * 128 + d];
        sV[i] = V[base + (size_t)tok * 128 + d];
    }
    __syncthreads();
    for (int q = tid; q < 512; q += 256) {
        float qv[16];
#pragma unroll
        for (int d = 0; d < 16; ++d) qv[d] = Q[base + (size_t)q * 128 + d];
        float mx = -1e30f, l = 0.0f, o[16];
#pragma unroll
        for (int d = 0; d < 16; ++d) o[d] = 0.0f;
        for (int j = 0; j < 512; ++j) {
            float s = 0.0f;
#pragma unroll
            for (int d = 0; d < 16; ++d) s += qv[d] * sK[j * 16 + d];
            s *= 0.25f;  // 1/sqrt(16)
            const float nm   = fmaxf(mx, s);
            const float corr = __expf(mx - nm);
            const float p    = __expf(s - nm);
            l = l * corr + p;
#pragma unroll
            for (int d = 0; d < 16; ++d) o[d] = o[d] * corr + p * sV[j * 16 + d];
            mx = nm;
        }
        const float inv = 1.0f / l;
#pragma unroll
        for (int d = 0; d < 16; ++d) O[base + (size_t)q * 128 + d] = o[d] * inv;
    }
}

// ---------------------------------------------------------------------------
static inline void gemm(hipStream_t st, const float* A, const float* A2,
                        const int* gis, const int* git, const float* B,
                        const float* bias, const float* mul, const float* res,
                        float* out, int M, int N, int K, int amode, int flags)
{
    dim3 grid((M + 127) / 128, N / 64);
    k_gemm<<<grid, dim3(256), 0, st>>>(A, A2, gis, git, B, bias, mul, res,
                                       out, M, N, K, amode, flags);
}

extern "C" void kernel_launch(void* const* d_in, const int* in_sizes, int n_in,
                              void* d_out, int out_size, void* d_ws, size_t ws_size,
                              hipStream_t stream)
{
    // ---- inputs (setup_inputs dict order; params in sorted-key pytree order)
    const float* a_x      = (const float*)d_in[0];
    const float* m_x      = (const float*)d_in[1];
    const float* m_e      = (const float*)d_in[2];
    const float* rbf3     = (const float*)d_in[3];
    const float* cbf3     = (const float*)d_in[4];
    const float* rbf_h    = (const float*)d_in[5];
    const float* a2m_w    = (const float*)d_in[6];
    const float* m2a_w    = (const float*)d_in[7];
    const float* a2m_attr = (const float*)d_in[8];
    const float* m2a_attr = (const float*)d_in[9];
    const float* W_atom   = (const float*)d_in[10];
    const float* W_ba     = (const float*)d_in[11];
    const float* W_cat    = (const float*)d_in[12];
    const float* W_cbf    = (const float*)d_in[13];
    const float* W_comb   = (const float*)d_in[14];
    const float* W_down   = (const float*)d_in[15];
    const float* W_rbf3   = (const float*)d_in[16];
    const float* W_rbfh   = (const float*)d_in[17];
    const float* W_up     = (const float*)d_in[18];
    const float* Wk       = (const float*)d_in[19];
    const float* Wo       = (const float*)d_in[20];
    const float* Wq       = (const float*)d_in[21];
    const float* Wv       = (const float*)d_in[22];
    const float* a2m_W1   = (const float*)d_in[23];
    const float* a2m_W2   = (const float*)d_in[24];
    const float* a2m_Wf1  = (const float*)d_in[25];
    const float* a2m_Wf2  = (const float*)d_in[26];
    const float* a2m_Win  = (const float*)d_in[27];
    const float* a2m_b1   = (const float*)d_in[28];
    const float* a2m_b2   = (const float*)d_in[29];
    const float* a2m_bf1  = (const float*)d_in[30];
    const float* a2m_bf2  = (const float*)d_in[31];
    const float* a2m_ln_b = (const float*)d_in[32];
    const float* a2m_ln_g = (const float*)d_in[33];
    const float* b_cat    = (const float*)d_in[34];
    const float* b_comb   = (const float*)d_in[35];
    const float* long_ln_b = (const float*)d_in[36];
    const float* long_ln_g = (const float*)d_in[37];
    const float* m2a_W1   = (const float*)d_in[38];
    const float* m2a_W2   = (const float*)d_in[39];
    const float* m2a_Wf1  = (const float*)d_in[40];
    const float* m2a_Wf2  = (const float*)d_in[41];
    const float* m2a_Win  = (const float*)d_in[42];
    const float* m2a_b1   = (const float*)d_in[43];
    const float* m2a_b2   = (const float*)d_in[44];
    const float* m2a_bf1  = (const float*)d_in[45];
    const float* m2a_bf2  = (const float*)d_in[46];
    const float* m2a_eln_b = (const float*)d_in[47];
    const float* m2a_eln_g = (const float*)d_in[48];
    const float* s_eln_b  = (const float*)d_in[49];
    const float* s_eln_g  = (const float*)d_in[50];
    const float* s_ln_b   = (const float*)d_in[51];
    const float* s_ln_g   = (const float*)d_in[52];
    // d_in[53] = id3_ragged_idx (unused by reference math)
    const int* id_swap = (const int*)d_in[54];
    const int* id3_ba  = (const int*)d_in[55];
    const int* id3_ca  = (const int*)d_in[56];
    const int* idx_s   = (const int*)d_in[57];
    const int* idx_t   = (const int*)d_in[58];
    const int* a2m_ei  = (const int*)d_in[59];  // [2, EA2M]: row0=src(atom), row1=dst(mesh)
    const int* m2a_ei  = (const int*)d_in[60];  // [2, EM2A]: row0=src(mesh), row1=dst(atom)

    // ---- outputs
    float* o_a = (float*)d_out;                          // a_n          [10000,128]
    float* o_m = o_a + (size_t)NATOMS * 128;             // mesh output  [32768,128]
    float* o_e = o_m + (size_t)NMESH * 128;              // edge output  [200000,256]

    // ---- workspace layout (floats)
    float* ws = (float*)d_ws;
    const size_t F_E256 = (size_t)NE * 256;
    const size_t NM128  = (size_t)NMESH * 128;
    const size_t EA128  = (size_t)EA2M * 128;
    float* wA   = ws;                       // [E,256]  r3 / x3 / gate / em_pre
    float* wB   = ws + F_E256;              // [E,256]  x_ba / m_new
    float* wD   = ws + 2 * F_E256;          // [E,64]   x_down
    float* wX3P = wD + (size_t)NE * 64;     // [E,64]   triplet aggregate
    float* wH   = wX3P + (size_t)NE * 64;   // [NA,256] h_pre
    // phase 2 (after short branch finishes) reuses wB.. region:
    float* wQ   = wB;
    float* wK   = wQ + NM128;
    float* wV   = wK + NM128;
    float* wAT  = wV + NM128;
    float* wMXO = wAT + NM128;
    float* wF1  = wMXO + NM128;             // [EA,128]
    float* wF2  = wF1 + EA128;              // [EA,128]
    float* wXIN = wF2 + EA128;              // [<=NM,128]
    float* wAGG = wXIN + NM128;             // [<=NM,128]
    float* wT   = wAGG + NM128;             // [<=NM,128]
    float* wMSG = wT + NM128;               // [NM,128] a2m message
    float* wMSG2 = wMSG + NM128;            // [NA,128] m2a message

    // ======================= short-range (GemNet triplet) ===================
    // m_ln -> o_e (scratch until m_out overwrites it); a_ln -> o_a
    k_ln<<<(NE + 7) / 8, 256, 0, stream>>>(m_e, s_eln_g, s_eln_b, o_e, NE, 256);
    k_ln<<<(NATOMS + 7) / 8, 256, 0, stream>>>(a_x, s_ln_g, s_ln_b, o_a, NATOMS, 128);
    // r3 = rbf3 @ W_rbf3
    gemm(stream, rbf3, nullptr, nullptr, nullptr, W_rbf3, nullptr, nullptr, nullptr,
         wA, NE, 256, 16, 0, 0);
    // x_ba = silu(m_ln @ W_ba) * r3
    gemm(stream, o_e, nullptr, nullptr, nullptr, W_ba, nullptr, wA, nullptr,
         wB, NE, 256, 256, 0, 2 | 4);
    // x_down = x_ba @ W_down
    gemm(stream, wB, nullptr, nullptr, nullptr, W_down, nullptr, nullptr, nullptr,
         wD, NE, 64, 256, 0, 0);
    // triplet scatter into x3pre
    k_zero<<<4096, 256, 0, stream>>>(wX3P, (long long)NE * 64);
    k_triplet<<<(NT + 3) / 4, 256, 0, stream>>>(cbf3, W_cbf, wD, id3_ba, id3_ca, wX3P, NT);
    // x3 = x3pre @ W_up
    gemm(stream, wX3P, nullptr, nullptr, nullptr, W_up, nullptr, nullptr, nullptr,
         wA, NE, 256, 64, 0, 0);
    // m_new = m_ln + x3 + x3[id_swap]
    k_mnew<<<NE, 256, 0, stream>>>(o_e, wA, id_swap, wB);
    // gate = rbf_h @ W_rbfh
    gemm(stream, rbf_h, nullptr, nullptr, nullptr, W_rbfh, nullptr, nullptr, nullptr,
         wA, NE, 256, 16, 0, 0);
    // h_pre[idx_t] += m_new * gate
    k_zero<<<4096, 256, 0, stream>>>(wH, (long long)NATOMS * 256);
    k_hagg<<<NE, 256, 0, stream>>>(wB, wA, idx_t, wH);
    // a_n = a_ln + silu(h_pre @ W_atom)   -> output 0
    gemm(stream, wH, nullptr, nullptr, nullptr, W_atom, nullptr, nullptr, o_a,
         o_a, NATOMS, 128, 256, 0, 2 | 8);
    // m_out = m_new + silu([a_n[idx_s], a_n[idx_t], m_new] @ W_cat + b_cat) -> output 2
    gemm(stream, o_a, wB, idx_s, idx_t, W_cat, b_cat, nullptr, wB,
         o_e, NE, 256, 640, 2, 1 | 2 | 8);

    // ======================= long-range (MHA) ==============================
    // mx_ln -> o_m (scratch until out2 overwrites it)
    k_ln<<<(NMESH + 7) / 8, 256, 0, stream>>>(m_x, long_ln_g, long_ln_b, o_m, NMESH, 128);
    gemm(stream, o_m, nullptr, nullptr, nullptr, Wq, nullptr, nullptr, nullptr,
         wQ, NMESH, 128, 128, 0, 0);
    gemm(stream, o_m, nullptr, nullptr, nullptr, Wk, nullptr, nullptr, nullptr,
         wK, NMESH, 128, 128, 0, 0);
    gemm(stream, o_m, nullptr, nullptr, nullptr, Wv, nullptr, nullptr, nullptr,
         wV, NMESH, 128, 128, 0, 0);
    k_attn<<<64 * 8, 256, 0, stream>>>(wQ, wK, wV, wAT);
    gemm(stream, wAT, nullptr, nullptr, nullptr, Wo, nullptr, nullptr, nullptr,
         wMXO, NMESH, 128, 128, 0, 0);

    // ======================= atom -> mesh CFConv ===========================
    gemm(stream, a2m_attr, nullptr, nullptr, nullptr, a2m_Wf1, a2m_bf1, nullptr, nullptr,
         wF1, EA2M, 128, 64, 0, 1 | 2);
    gemm(stream, wF1, nullptr, nullptr, nullptr, a2m_Wf2, a2m_bf2, nullptr, nullptr,
         wF2, EA2M, 128, 128, 0, 1);
    gemm(stream, o_a, nullptr, nullptr, nullptr, a2m_Win, nullptr, nullptr, nullptr,
         wXIN, NATOMS, 128, 128, 0, 0);
    k_zero<<<4096, 256, 0, stream>>>(wAGG, (long long)NMESH * 128);
    k_cfconv<<<(EA2M * 128) / 256, 256, 0, stream>>>(wXIN, wF2, a2m_w,
                                                     a2m_ei, a2m_ei + EA2M, wAGG, EA2M);
    gemm(stream, wAGG, nullptr, nullptr, nullptr, a2m_W1, a2m_b1, nullptr, nullptr,
         wT, NMESH, 128, 128, 0, 1 | 2);
    gemm(stream, wT, nullptr, nullptr, nullptr, a2m_W2, a2m_b2, nullptr, nullptr,
         wMSG, NMESH, 128, 128, 0, 1);
    // output 1 = m_x + mx + LN(a2m_msg)
    k_out2<<<(NMESH + 7) / 8, 256, 0, stream>>>(wMSG, m_x, wMXO, a2m_ln_g, a2m_ln_b,
                                                o_m, NMESH);

    // ======================= mesh -> atom CFConv + edge lift ===============
    gemm(stream, m2a_attr, nullptr, nullptr, nullptr, m2a_Wf1, m2a_bf1, nullptr, nullptr,
         wF1, EA2M, 128, 64, 0, 1 | 2);
    gemm(stream, wF1, nullptr, nullptr, nullptr, m2a_Wf2, m2a_bf2, nullptr, nullptr,
         wF2, EA2M, 128, 128, 0, 1);
    gemm(stream, wMXO, nullptr, nullptr, nullptr, m2a_Win, nullptr, nullptr, nullptr,
         wXIN, NMESH, 128, 128, 0, 0);
    k_zero<<<4096, 256, 0, stream>>>(wAGG, (long long)NATOMS * 128);
    k_cfconv<<<(EA2M * 128) / 256, 256, 0, stream>>>(wXIN, wF2, m2a_w,
                                                     m2a_ei, m2a_ei + EA2M, wAGG, EA2M);
    gemm(stream, wAGG, nullptr, nullptr, nullptr, m2a_W1, m2a_b1, nullptr, nullptr,
         wT, NATOMS, 128, 128, 0, 1 | 2);
    gemm(stream, wT, nullptr, nullptr, nullptr, m2a_W2, m2a_b2, nullptr, nullptr,
         wMSG2, NATOMS, 128, 128, 0, 1);
    // em_pre = silu([msg[idx_s], msg[idx_t]] @ W_comb + b_comb)
    gemm(stream, wMSG2, nullptr, idx_s, idx_t, W_comb, b_comb, nullptr, nullptr,
         wA, NE, 256, 256, 1, 1 | 2);
    // output 2 += LN(em_pre)
    k_out3<<<(NE + 7) / 8, 256, 0, stream>>>(wA, m2a_eln_g, m2a_eln_b, o_e, NE);
}
